// AttentionProbe_14061722927109
// MI455X (gfx1250) — matrix operand order repaired and checked by reading, not compiled, not run on hardware
//
#include <hip/hip_runtime.h>

// ---------------------------------------------------------------------------
// AttentionProbe for MI455X (gfx1250, wave32, WMMA, TDM + async-LDS copies).
// Pipeline:
//   K0a: x (f32) -> xb (bf16)
//   K0b: W_qkv (f32, KxN) -> Wt (bf16, NxK transposed)
//   K1 : qkv = x @ W_qkv + b   bf16 WMMA, f32 accum; tiles streamed by the
//        Tensor Data Mover (tensor_load_to_lds, double-buffered LDS,
//        s_wait_tensorcnt), 8 WMMA per wave per K-step.
//   K2 : sliding-window causal flash attention (WMMA scores + WMMA P@V,
//        online softmax), K staged via global_load_async_to_lds_b128.
//   K3 : out = attn_out @ W_out + b_out  (wave32 shuffle reduction)
// ---------------------------------------------------------------------------

#define BATCH   4
#define SEQ     2048
#define DMODEL  2048
#define NHEAD   8
#define DPROJ   64
#define NQKV    (3 * NHEAD * DPROJ)   // 1536
#define WINDOW  256

typedef __attribute__((ext_vector_type(16))) __bf16 v16bf;
typedef __attribute__((ext_vector_type(8)))  __bf16 bf16x8;
typedef __attribute__((ext_vector_type(2)))  __bf16 bf16x2;
typedef __attribute__((ext_vector_type(8)))  float  v8f;
typedef __attribute__((ext_vector_type(4)))  unsigned int v4u;
typedef __attribute__((ext_vector_type(8)))  int v8i;
typedef __attribute__((ext_vector_type(4)))  int v4i;

union FragB16 { v16bf v; bf16x8 h[2]; };

// A-fragment (16x32 16-bit, ISA 7.12.2): two contiguous 16B chunks per lane.
__device__ __forceinline__ v16bf load_a_frag(const __bf16* p, int half) {
    FragB16 f;
    f.h[0] = *(const bf16x8*)(p + 8 * half);
    f.h[1] = *(const bf16x8*)(p + 16 + 8 * half);
    return f.v;
}
// B-fragment (32x16 16-bit): two contiguous 16B chunks per lane, same k order as A.
__device__ __forceinline__ v16bf load_b_frag(const __bf16* p, int half) {
    FragB16 f;
    f.h[0] = *(const bf16x8*)(p + 8 * half);
    f.h[1] = *(const bf16x8*)(p + 16 + 8 * half);
    return f.v;
}

// Async 16B global->LDS copy (ASYNCcnt-tracked).
__device__ __forceinline__ void async_b128(const __bf16* gp, const __bf16* lp) {
    unsigned lds = (unsigned)(unsigned long long)lp;  // low 32 bits = LDS offset
    asm volatile("global_load_async_to_lds_b128 %0, %1, off"
                 :: "v"(lds), "v"(gp) : "memory");
}
__device__ __forceinline__ void wait_async0() {
    asm volatile("s_wait_asynccnt 0" ::: "memory");
}

// ---------------------------------------------------------------------------
// Tensor Data Mover: one descriptor loads a (tile_rows x 32) bf16 tile whose
// source rows are DMODEL apart, into LDS rows padded to 40 bf16 (80B).
// D# per ISA sec.8: group0 = {count=1 | lds_addr | global_addr(57b) | type=2},
// group1 = {data_size=2B, pad_enable, pad_interval=64B, pad_amount=16B,
//           tensor_dim0/1, tile_dim0=32, tile_dim1, dim0_stride}.
// This toolchain's builtin takes 6 args (g0, g1, g2, g3, g4, cpol).
// ---------------------------------------------------------------------------
__device__ __forceinline__ void tdm_load_tile(const __bf16* gp, const __bf16* lp,
                                              unsigned tensor_rows,
                                              unsigned tile_rows)
{
    unsigned lds = (unsigned)(unsigned long long)lp;
    unsigned long long ga = (unsigned long long)gp;
    v4u g0;
    g0[0] = 1u;                                        // count=1 (valid), user
    g0[1] = lds;                                       // lds_addr (bytes)
    g0[2] = (unsigned)ga;                              // global_addr[31:0]
    g0[3] = (unsigned)((ga >> 32) & 0x1FFFFFFu)        // global_addr[56:32]
            | (2u << 30);                              // type=2 ("image")
    v8i g1;
    g1[0] = (int)((1u << 16)        // data_size = 1 -> 2 bytes
                | (1u << 20)        // pad_enable
                | (3u << 22)        // pad_interval: 16 DWORDs = 64B
                | (3u << 25));      // pad_amount:   4 DWORDs = 16B
    g1[1] = (int)(((unsigned)DMODEL & 0xFFFFu) << 16);            // dim0 lo16
    g1[2] = (int)(((unsigned)DMODEL >> 16) | ((tensor_rows & 0xFFFFu) << 16));
    g1[3] = (int)((tensor_rows >> 16) | (32u << 16));             // tile_dim0=32
    g1[4] = (int)tile_rows;                                       // tile_dim1
    g1[5] = (int)DMODEL;                                          // dim0 stride
    g1[6] = 0;
    g1[7] = 0;
    v4i z4 = (v4i){0, 0, 0, 0};
    v8i z8 = (v8i){0, 0, 0, 0, 0, 0, 0, 0};
    __builtin_amdgcn_tensor_load_to_lds(g0, g1, z4, z4, z8, 0);
}

// ---------------------------------------------------------------------------
// K0a: f32 -> bf16 elementwise (x).
// ---------------------------------------------------------------------------
__global__ __launch_bounds__(256)
void convert_x_bf16(const float* __restrict__ x, __bf16* __restrict__ xb)
{
    int i = (blockIdx.x * 256 + threadIdx.x) * 8;
    const float4* x4 = (const float4*)(x + i);
    float4 a = x4[0], b = x4[1];
    FragB16 f;
    f.h[0][0] = (__bf16)a.x; f.h[0][1] = (__bf16)a.y;
    f.h[0][2] = (__bf16)a.z; f.h[0][3] = (__bf16)a.w;
    f.h[0][4] = (__bf16)b.x; f.h[0][5] = (__bf16)b.y;
    f.h[0][6] = (__bf16)b.z; f.h[0][7] = (__bf16)b.w;
    *(bf16x8*)(xb + i) = f.h[0];
}

// ---------------------------------------------------------------------------
// K0b: W_qkv [K=2048][N=1536] f32 -> Wt [N][K] bf16 (LDS tile transpose).
// ---------------------------------------------------------------------------
__global__ __launch_bounds__(256)
void convert_w_bf16(const float* __restrict__ W, __bf16* __restrict__ Wt)
{
    __shared__ float tile[32][33];
    const int k0 = blockIdx.x * 32, n0 = blockIdx.y * 32;
    const int c = threadIdx.x & 31, r = threadIdx.x >> 5;
    for (int rr = r; rr < 32; rr += 8)
        tile[rr][c] = W[(k0 + rr) * NQKV + n0 + c];
    __syncthreads();
    for (int rr = r; rr < 32; rr += 8)
        Wt[(n0 + rr) * DMODEL + k0 + c] = (__bf16)tile[c][rr];
}

// ---------------------------------------------------------------------------
// K1: QKV GEMM.  M=8192, N=1536, K=2048.  Block tile 128x128, 8 waves
// (4Mx2N), wave tile 32x64 (2x4 WMMA).  TDM-streamed, double-buffered LDS.
// ---------------------------------------------------------------------------
__global__ __launch_bounds__(256)
void qkv_gemm_wmma(const __bf16* __restrict__ xb,
                   const __bf16* __restrict__ Wt,
                   const float* __restrict__ bqkv,
                   __bf16* __restrict__ qb,
                   __bf16* __restrict__ kb,
                   __bf16* __restrict__ vb)
{
    __shared__ alignas(16) __bf16 As[2][128][40];   // [buf][m][k], 80B rows
    __shared__ alignas(16) __bf16 Bt[2][128][40];   // [buf][n][k]

    const int tid  = threadIdx.x;
    const int lane = tid & 31;
    const int wave = tid >> 5;
    const int ln   = lane & 15;
    const int half = lane >> 4;
    const int wm   = wave & 3;       // 4 waves along M (32 rows each)
    const int wn   = wave >> 2;      // 2 waves along N (64 cols each)
    const int m0   = blockIdx.x * 128;
    const int n0   = blockIdx.y * 128;

    v8f acc[2][4];
    for (int i = 0; i < 2; ++i)
        for (int j = 0; j < 4; ++j)
            acc[i][j] = (v8f){0.f,0.f,0.f,0.f,0.f,0.f,0.f,0.f};

    // Prime buffer 0 (wave 0 drives the Tensor Data Mover).
    if (tid < 32) {
        tdm_load_tile(xb + (size_t)m0 * DMODEL, &As[0][0][0], BATCH * SEQ, 128);
        tdm_load_tile(Wt + (size_t)n0 * DMODEL, &Bt[0][0][0], NQKV, 128);
    }

    for (int k0 = 0; k0 < DMODEL; k0 += 32) {
        const int cur = (k0 >> 5) & 1;
        if (tid < 32)
            __builtin_amdgcn_s_wait_tensorcnt(0);
        __syncthreads();   // DMA of `cur` visible to all; prior reads of cur^1 done
        if (tid < 32 && k0 + 32 < DMODEL) {
            tdm_load_tile(xb + (size_t)m0 * DMODEL + k0 + 32,
                          &As[cur ^ 1][0][0], BATCH * SEQ, 128);
            tdm_load_tile(Wt + (size_t)n0 * DMODEL + k0 + 32,
                          &Bt[cur ^ 1][0][0], NQKV, 128);
        }

        v16bf afr[2], bfr[4];
        afr[0] = load_a_frag(&As[cur][wm * 32 + ln][0],      half);
        afr[1] = load_a_frag(&As[cur][wm * 32 + 16 + ln][0], half);
        #pragma unroll
        for (int ns = 0; ns < 4; ++ns)
            bfr[ns] = load_b_frag(&Bt[cur][wn * 64 + ns * 16 + ln][0], half);

        #pragma unroll
        for (int ms = 0; ms < 2; ++ms)
            #pragma unroll
            for (int ns = 0; ns < 4; ++ns)
                acc[ms][ns] = __builtin_amdgcn_wmma_f32_16x16x32_bf16(
                    false, afr[ms], false, bfr[ns], (short)0, acc[ms][ns],
                    false, false);
    }

    // Epilogue: +bias, scatter into q/k/v [B][H][S][DPROJ] bf16.
    for (int ms = 0; ms < 2; ++ms) {
        for (int ns = 0; ns < 4; ++ns) {
            int n   = n0 + wn * 64 + ns * 16 + ln;
            float bv = bqkv[n];
            int sel = n >> 9;
            int h   = (n >> 6) & 7;
            int d   = n & 63;
            __bf16* dst = (sel == 0) ? qb : (sel == 1) ? kb : vb;
            #pragma unroll
            for (int e = 0; e < 8; ++e) {
                int m = m0 + wm * 32 + ms * 16 + e + 8 * half;
                int s = m & (SEQ - 1);
                int bidx = m >> 11;
                dst[((bidx * NHEAD + h) * SEQ + s) * DPROJ + d] =
                    (__bf16)(acc[ms][ns][e] + bv);
            }
        }
    }
}

// ---------------------------------------------------------------------------
// K2: sliding-window causal flash attention, one (b,h,64-query) tile/block.
// ---------------------------------------------------------------------------
#define NCHUNK 10   // keys [q0-256, q0+63] in 10 chunks of 32

__global__ __launch_bounds__(128)
void attn_wmma(const __bf16* __restrict__ qb,
               const __bf16* __restrict__ kb,
               const __bf16* __restrict__ vb,
               const unsigned char* __restrict__ pmask,
               float* __restrict__ attn_out)
{
    __shared__ alignas(16) __bf16 Kt[32][72];      // [key][d], 144B rows
    __shared__ alignas(16) __bf16 Vt[64][40];      // [d][key]
    __shared__ alignas(16) __bf16 Pw[4][16][40];   // per-wave probs [q][key]
    __shared__ float Sc[4][16][32];
    __shared__ float corr_s[4][16];
    __shared__ float linv_s[4][16];

    const int tid  = threadIdx.x;
    const int lane = tid & 31;
    const int wave = tid >> 5;
    const int ln   = lane & 15;
    const int half = lane >> 4;

    const int q0 = blockIdx.x * 64;
    const int bh = blockIdx.y;
    const int b  = bh / NHEAD;
    const int h  = bh % NHEAD;
    const int qbase = q0 + wave * 16;
    const int head_base = bh * SEQ * DPROJ;

    v16bf aq[2];
    aq[0] = load_a_frag(qb + head_base + (qbase + ln) * DPROJ,      half);
    aq[1] = load_a_frag(qb + head_base + (qbase + ln) * DPROJ + 32, half);

    v8f acc[4];
    for (int i = 0; i < 4; ++i)
        acc[i] = (v8f){0.f,0.f,0.f,0.f,0.f,0.f,0.f,0.f};
    float row_m = -1e30f;
    float row_l = 0.0f;

    for (int c = 0; c < NCHUNK; ++c) {
        const int ks = q0 - WINDOW + 32 * c;

        // K chunk 32x64 bf16 via async b128 (clamped addr; masked later).
        for (int c2 = tid; c2 < 256; c2 += 128) {
            int key = c2 >> 3, seg = c2 & 7;
            int kg = ks + key;
            kg = kg < 0 ? 0 : (kg >= SEQ ? SEQ - 1 : kg);
            async_b128(kb + head_base + (size_t)kg * DPROJ + seg * 8,
                       &Kt[key][seg * 8]);
        }
        // V chunk, transposed + zero-filled through VGPRs.
        for (int i = tid; i < 32 * 64; i += 128) {
            int key = i >> 6, d = i & 63;
            int kg = ks + key;
            __bf16 vv = (__bf16)0.f;
            if (kg >= 0 && kg < SEQ)
                vv = vb[head_base + kg * DPROJ + d];
            Vt[d][key] = vv;
        }
        wait_async0();
        __syncthreads();

        for (int keysub = 0; keysub < 2; ++keysub) {
            v8f sacc = (v8f){0.f,0.f,0.f,0.f,0.f,0.f,0.f,0.f};
            #pragma unroll
            for (int kstep = 0; kstep < 2; ++kstep) {
                v16bf bk = load_b_frag(&Kt[keysub * 16 + ln][kstep * 32], half);
                sacc = __builtin_amdgcn_wmma_f32_16x16x32_bf16(
                    false, aq[kstep], false, bk, (short)0, sacc, false, false);
            }
            #pragma unroll
            for (int e = 0; e < 8; ++e) {
                int m  = e + 8 * half;
                int qg = qbase + m;
                int kg = ks + keysub * 16 + ln;
                bool ok = (kg >= 0) && (kg <= qg) && (qg - kg < WINDOW) &&
                          (pmask[b * SEQ + (kg < 0 ? 0 : kg)] != 0);
                Sc[wave][m][keysub * 16 + ln] = ok ? sacc[e] * 0.125f : -1e30f;
            }
        }
        __syncthreads();

        if (lane < 16) {
            float rmax = -1e30f;
            #pragma unroll 4
            for (int j = 0; j < 32; ++j) rmax = fmaxf(rmax, Sc[wave][lane][j]);
            float mnew = fmaxf(row_m, rmax);
            float corr = __expf(row_m - mnew);
            float sum = 0.f;
            #pragma unroll 4
            for (int j = 0; j < 32; j += 2) {
                float s0 = Sc[wave][lane][j];
                float s1 = Sc[wave][lane][j + 1];
                float p0 = (s0 > -5e29f) ? __expf(s0 - mnew) : 0.f;
                float p1 = (s1 > -5e29f) ? __expf(s1 - mnew) : 0.f;
                bf16x2 pp; pp.x = (__bf16)p0; pp.y = (__bf16)p1;
                *(bf16x2*)&Pw[wave][lane][j] = pp;
                sum += p0 + p1;
            }
            row_l = row_l * corr + sum;
            row_m = mnew;
            corr_s[wave][lane] = corr;
        }
        __syncthreads();

        float ce[8];
        #pragma unroll
        for (int e = 0; e < 8; ++e) ce[e] = corr_s[wave][e + 8 * half];
        #pragma unroll
        for (int dsub = 0; dsub < 4; ++dsub)
            #pragma unroll
            for (int e = 0; e < 8; ++e) acc[dsub][e] *= ce[e];

        v16bf pa = load_a_frag(&Pw[wave][ln][0], half);
        #pragma unroll
        for (int dsub = 0; dsub < 4; ++dsub) {
            v16bf bv = load_b_frag(&Vt[dsub * 16 + ln][0], half);
            acc[dsub] = __builtin_amdgcn_wmma_f32_16x16x32_bf16(
                false, pa, false, bv, (short)0, acc[dsub], false, false);
        }
        __syncthreads();
    }

    if (lane < 16)
        linv_s[wave][lane] = (row_l > 0.f) ? (1.0f / row_l) : 0.f;
    __syncthreads();
    float le[8];
    #pragma unroll
    for (int e = 0; e < 8; ++e) le[e] = linv_s[wave][e + 8 * half];
    #pragma unroll
    for (int dsub = 0; dsub < 4; ++dsub) {
        #pragma unroll
        for (int e = 0; e < 8; ++e) {
            int srow = qbase + e + 8 * half;
            attn_out[(b * SEQ + srow) * (NHEAD * DPROJ) + h * DPROJ +
                     dsub * 16 + ln] = acc[dsub][e] * le[e];
        }
    }
}

// ---------------------------------------------------------------------------
// K3: out = attn_out @ W_out + b_out.  One wave per output row.
// ---------------------------------------------------------------------------
__global__ __launch_bounds__(256)
void out_proj(const float* __restrict__ attn_out,
              const float* __restrict__ Wout,
              const float* __restrict__ bout,
              float* __restrict__ out)
{
    const int row  = blockIdx.x * 8 + (threadIdx.x >> 5);
    const int lane = threadIdx.x & 31;
    const float* a = attn_out + row * (NHEAD * DPROJ);
    float s = 0.f;
    #pragma unroll
    for (int j = lane; j < NHEAD * DPROJ; j += 32)
        s += a[j] * Wout[j];
    #pragma unroll
    for (int off = 16; off > 0; off >>= 1)
        s += __shfl_down(s, off, 32);
    if (lane == 0) out[row] = s + bout[0];
}

// ---------------------------------------------------------------------------
extern "C" void kernel_launch(void* const* d_in, const int* in_sizes, int n_in,
                              void* d_out, int out_size, void* d_ws, size_t ws_size,
                              hipStream_t stream)
{
    const float*         x    = (const float*)d_in[0];
    const unsigned char* pm   = (const unsigned char*)d_in[1];
    const float*         Wqkv = (const float*)d_in[2];
    const float*         bqkv = (const float*)d_in[3];
    const float*         Wout = (const float*)d_in[4];
    const float*         bout = (const float*)d_in[5];
    float*               out  = (float*)d_out;

    char* ws = (char*)d_ws;
    const size_t MB = 1024 * 1024;
    __bf16* qb   = (__bf16*)(ws);                 //  8 MB
    __bf16* kb   = (__bf16*)(ws +  8 * MB);       //  8 MB
    __bf16* vb   = (__bf16*)(ws + 16 * MB);       //  8 MB
    float*  attn = (float*)(ws + 24 * MB);        // 16 MB
    __bf16* xb   = (__bf16*)(ws + 40 * MB);       // 32 MB (8192 x 2048)
    __bf16* Wt   = (__bf16*)(ws + 72 * MB);       //  6 MB (1536 x 2048)

    (void)in_sizes; (void)n_in; (void)out_size; (void)ws_size;

    convert_x_bf16<<<(BATCH * SEQ * DMODEL) / (256 * 8), 256, 0, stream>>>(x, xb);
    convert_w_bf16<<<dim3(DMODEL / 32, NQKV / 32), 256, 0, stream>>>(Wqkv, Wt);
    qkv_gemm_wmma<<<dim3(64, 12), 256, 0, stream>>>(xb, Wt, bqkv, qb, kb, vb);
    attn_wmma<<<dim3(SEQ / 64, BATCH * NHEAD), 128, 0, stream>>>(qb, kb, vb, pm, attn);
    out_proj<<<(BATCH * SEQ) / 8, 256, 0, stream>>>(attn, Wout, bout, out);
}
